// EdgeRankingGNN2_Ablation1_41875931136405
// MI455X (gfx1250) — matrix operand order repaired and checked by reading, not compiled, not run on hardware
//
#include <hip/hip_runtime.h>
#include <hip/hip_bf16.h>
#include <math.h>

// ---------------------------------------------------------------------------
// CDNA5 (gfx1250) WMMA types & fragment helpers (wave32 layouts, ISA 7.12.2)
// ---------------------------------------------------------------------------
typedef __attribute__((ext_vector_type(16))) __bf16 v16bf;
typedef __attribute__((ext_vector_type(8)))  __bf16 v8bf;
typedef __attribute__((ext_vector_type(8)))  float  v8f;

static __device__ inline v8f vzero8() {
  v8f z;
#pragma unroll
  for (int i = 0; i < 8; ++i) z[i] = 0.0f;
  return z;
}

// A-matrix 16x32 bf16 fragment from LDS (row-major tile, stride lda elements).
// Lane l in [0,16): row M=l. Elements 0..7 -> K = hi*8+0..7 (contiguous),
// elements 8..15 -> K = 16+hi*8+0..7 (contiguous). Two aligned 16B loads.
static __device__ inline v16bf load_a_frag(const __bf16* base, int lda, int l, int hi) {
  const __bf16* row = base + l * lda;
  v8bf lo = *(const v8bf*)(row + hi * 8);
  v8bf hh = *(const v8bf*)(row + 16 + hi * 8);
  return __builtin_shufflevector(lo, hh, 0, 1, 2, 3, 4, 5, 6, 7,
                                 8, 9, 10, 11, 12, 13, 14, 15);
}

// ---------------------------------------------------------------------------
// Kernel: first layer of 3->64 encoders: out = relu(x @ w1 + b1)
// ---------------------------------------------------------------------------
__global__ __launch_bounds__(256) void encode3_kernel(
    const float* __restrict__ x, const float* __restrict__ w1,
    const float* __restrict__ b1, float* __restrict__ out, int R) {
  __shared__ float sw[3 * 64];
  __shared__ float sb[64];
  int tid = threadIdx.x;
  if (tid < 192) sw[tid] = w1[tid];
  if (tid < 64) sb[tid] = b1[tid];
  __syncthreads();
  long long idx = (long long)blockIdx.x * blockDim.x + tid;
  if (idx >= (long long)R * 64) return;
  int r = (int)(idx >> 6), c = (int)(idx & 63);
  const float* xr = x + (long long)r * 3;
  float v = sb[c] + xr[0] * sw[c] + xr[1] * sw[64 + c] + xr[2] * sw[128 + c];
  out[idx] = fmaxf(v, 0.0f);
}

// ---------------------------------------------------------------------------
// Kernel: fused [R,64] @ [64,64] + bias (+combine, +relu, +LayerNorm, +relu)
// FLAGS: bit0 = relu (before LN if LN set), bit1 = LN, bit2 = relu after LN,
//        bit3 = A := (1+eps)*A + A2
// Block = 128 threads (4 waves), 64 rows per block. WMMA bf16, fp32 accum.
// W is staged in LDS pre-swizzled to fragment order: lWs[kk][nt][lane][e],
// so each lane's B fragment is one contiguous 32B vector load.
// ---------------------------------------------------------------------------
template <int FLAGS>
__global__ __launch_bounds__(128) void gemm64_kernel(
    const float* __restrict__ A, const float* __restrict__ A2,
    const float* __restrict__ epsp, const float* __restrict__ W,
    const float* __restrict__ bias, const float* __restrict__ g,
    const float* __restrict__ be, float* __restrict__ out, int R) {
  __shared__ __align__(16) __bf16 lA[64][72];
  __shared__ __align__(32) __bf16 lWs[2][4][32][16];  // [kTile][nTile][lane][elem]
  __shared__ float lO[64][65];

  const int tid = threadIdx.x;
  const long long rowBlock = (long long)blockIdx.x * 64;
  float eps1 = 1.0f;
  if (FLAGS & 8) eps1 = 1.0f + epsp[0];

  // Stage A (with optional GIN combine) as bf16.
  for (int i = tid; i < 64 * 64; i += 128) {
    int r = i >> 6, c = i & 63;
    long long gr = rowBlock + r;
    float v = 0.0f;
    if (gr < R) {
      v = A[gr * 64 + c];
      if (FLAGS & 8) v = eps1 * v + A2[gr * 64 + c];
    }
    lA[r][c] = (__bf16)v;
  }
  // Stage W pre-swizzled: fragment element e of lane ln, tile (kk,nt)
  //   corresponds to W[kk*32 + (ln>>4)*8 + (e&7) + (e>>3)*16][nt*16 + (ln&15)].
  for (int i = tid; i < 64 * 64; i += 128) {
    int e = i & 15, ln = (i >> 4) & 31, nt = (i >> 9) & 3, kk = i >> 11;
    int k = kk * 32 + (ln >> 4) * 8 + (e & 7) + (e >> 3) * 16;
    int n = nt * 16 + (ln & 15);
    lWs[kk][nt][ln][e] = (__bf16)W[k * 64 + n];
  }
  __syncthreads();

  const int wave = tid >> 5;
  const int lane = tid & 31;
  const int l = lane & 15, hi = lane >> 4;

  v8f acc[4];
#pragma unroll
  for (int nt = 0; nt < 4; ++nt) acc[nt] = vzero8();

#pragma unroll
  for (int kk = 0; kk < 2; ++kk) {
    v16bf a = load_a_frag(&lA[wave * 16][kk * 32], 72, l, hi);
#pragma unroll
    for (int nt = 0; nt < 4; ++nt) {
      v16bf b = *(const v16bf*)&lWs[kk][nt][lane][0];
      acc[nt] = __builtin_amdgcn_wmma_f32_16x16x32_bf16(
          false, a, false, b, (short)0, acc[nt], false, false);
    }
  }

#pragma unroll
  for (int nt = 0; nt < 4; ++nt)
#pragma unroll
    for (int r = 0; r < 8; ++r)
      lO[wave * 16 + hi * 8 + r][nt * 16 + l] = acc[nt][r] + bias[nt * 16 + l];
  __syncthreads();

  if (tid < 64) {
    long long gr = rowBlock + tid;
    if (gr < R) {
      float* orow = out + gr * 64;
      if (FLAGS & 2) {
        float m = 0.0f;
        for (int c = 0; c < 64; ++c) {
          float v = lO[tid][c];
          if (FLAGS & 1) v = fmaxf(v, 0.0f);
          lO[tid][c] = v;
          m += v;
        }
        m *= (1.0f / 64.0f);
        float s2 = 0.0f;
        for (int c = 0; c < 64; ++c) {
          float d = lO[tid][c] - m;
          s2 += d * d;
        }
        float inv = rsqrtf(s2 * (1.0f / 64.0f) + 1e-5f);
        for (int c = 0; c < 64; ++c) {
          float y = (lO[tid][c] - m) * inv * g[c] + be[c];
          if (FLAGS & 4) y = fmaxf(y, 0.0f);
          orow[c] = y;
        }
      } else {
        for (int c = 0; c < 64; ++c) {
          float v = lO[tid][c];
          if (FLAGS & 1) v = fmaxf(v, 0.0f);
          orow[c] = v;
        }
      }
    }
  }
}

// ---------------------------------------------------------------------------
// Kernel: edge-predictor GEMM, bf16 A in global, W f32, tanh, bf16 out.
// out[R,N] = tanh(A[R,K] @ W[K,N] + bias). K streamed in 32-wide slices,
// W slice staged pre-swizzled per fragment (one 32B vector load per B frag).
// ---------------------------------------------------------------------------
template <int K, int N>
__global__ __launch_bounds__(128) void gemm_big_kernel(
    const __bf16* __restrict__ A, const float* __restrict__ W,
    const float* __restrict__ bias, __bf16* __restrict__ out, int R) {
  __shared__ __align__(16) __bf16 lA[64][40];
  __shared__ __align__(32) __bf16 lWs[N / 16][32][16];  // [nTile][lane][elem]

  const int tid = threadIdx.x;
  const long long rowBlock = (long long)blockIdx.x * 64;
  const int wave = tid >> 5, lane = tid & 31;
  const int l = lane & 15, hi = lane >> 4;

  v8f acc[N / 16];
#pragma unroll
  for (int nt = 0; nt < N / 16; ++nt) acc[nt] = vzero8();

  for (int k0 = 0; k0 < K; k0 += 32) {
    for (int i = tid; i < 64 * 32; i += 128) {
      int r = i >> 5, c = i & 31;
      long long gr = rowBlock + r;
      lA[r][c] = (gr < R) ? A[gr * K + k0 + c] : (__bf16)0.0f;
    }
    for (int i = tid; i < 32 * N; i += 128) {
      int e = i & 15, ln = (i >> 4) & 31, nt = i >> 9;
      int k = (ln >> 4) * 8 + (e & 7) + (e >> 3) * 16;
      int n = nt * 16 + (ln & 15);
      lWs[nt][ln][e] = (__bf16)W[(long long)(k0 + k) * N + n];
    }
    __syncthreads();

    v16bf a = load_a_frag(&lA[wave * 16][0], 40, l, hi);
#pragma unroll
    for (int nt = 0; nt < N / 16; ++nt) {
      v16bf b = *(const v16bf*)&lWs[nt][lane][0];
      acc[nt] = __builtin_amdgcn_wmma_f32_16x16x32_bf16(
          false, a, false, b, (short)0, acc[nt], false, false);
    }
    __syncthreads();
  }

#pragma unroll
  for (int nt = 0; nt < N / 16; ++nt) {
#pragma unroll
    for (int r = 0; r < 8; ++r) {
      int row = wave * 16 + hi * 8 + r;
      long long gr = rowBlock + row;
      if (gr < R) {
        int col = nt * 16 + l;
        out[gr * N + col] = (__bf16)tanhf(acc[nt][r] + bias[col]);
      }
    }
  }
}

// ---------------------------------------------------------------------------
// Irregular kernels
// ---------------------------------------------------------------------------
__global__ __launch_bounds__(256) void zero_kernel(float* __restrict__ p, long long n) {
  long long i = (long long)blockIdx.x * blockDim.x + threadIdx.x;
  if (i < n) p[i] = 0.0f;
}

// aggr[dst] += relu(h[src] + ef)  (one thread per edge-channel)
__global__ __launch_bounds__(256) void scatter_msg_kernel(
    const float* __restrict__ h, const float* __restrict__ ef,
    const int* __restrict__ src, const int* __restrict__ dst,
    float* __restrict__ aggr, int E) {
  long long idx = (long long)blockIdx.x * blockDim.x + threadIdx.x;
  if (idx >= (long long)E * 64) return;
  int e = (int)(idx >> 6), c = (int)(idx & 63);
  int s = src[e], d = dst[e];
  float v = h[(long long)s * 64 + c] + ef[idx];
  if (v > 0.0f) atomicAdd(&aggr[(long long)d * 64 + c], v);
}

__global__ __launch_bounds__(256) void pool_scatter_kernel(
    const float* __restrict__ h, const int* __restrict__ batch,
    float* __restrict__ seg, float* __restrict__ cnt, int Nn) {
  long long idx = (long long)blockIdx.x * blockDim.x + threadIdx.x;
  if (idx >= (long long)Nn * 64) return;
  int n = (int)(idx >> 6), c = (int)(idx & 63);
  int b = batch[n];
  atomicAdd(&seg[b * 64 + c], h[idx]);
  if (c == 0) atomicAdd(&cnt[b], 1.0f);
}

__global__ __launch_bounds__(256) void pool_finalize_kernel(
    const float* __restrict__ seg, const float* __restrict__ cnt,
    float* __restrict__ gp) {
  int idx = blockIdx.x * blockDim.x + threadIdx.x;
  if (idx >= 64 * 64) return;
  gp[idx] = seg[idx] / fmaxf(cnt[idx >> 6], 1.0f);
}

// C[e] = concat(h[src[e]], h[dst[e]], gf[batch[src[e]]], ef[e]) as bf16 [E,256]
__global__ __launch_bounds__(256) void concat_kernel(
    const float* __restrict__ h, const float* __restrict__ ef,
    const float* __restrict__ gf, const int* __restrict__ src,
    const int* __restrict__ dst, const int* __restrict__ batch,
    __bf16* __restrict__ C, int E) {
  long long idx = (long long)blockIdx.x * blockDim.x + threadIdx.x;
  if (idx >= (long long)E * 4) return;
  int e = (int)(idx >> 2), seg = (int)(idx & 3);
  int s = src[e];
  const float* p;
  if (seg == 0)      p = h + (long long)s * 64;
  else if (seg == 1) p = h + (long long)dst[e] * 64;
  else if (seg == 2) p = gf + (long long)batch[s] * 64;
  else               p = ef + (long long)e * 64;
  __bf16* o = C + (long long)e * 256 + seg * 64;
#pragma unroll 4
  for (int c = 0; c < 64; ++c) o[c] = (__bf16)p[c];
}

// out[e] = sigmoid(t2[e,:] . w3 + b3)
__global__ __launch_bounds__(256) void final_kernel(
    const __bf16* __restrict__ t2, const float* __restrict__ w3,
    const float* __restrict__ b3, float* __restrict__ out, int E) {
  __shared__ float sw[64];
  if (threadIdx.x < 64) sw[threadIdx.x] = w3[threadIdx.x];
  __syncthreads();
  long long e = (long long)blockIdx.x * blockDim.x + threadIdx.x;
  if (e >= E) return;
  const __bf16* t = t2 + e * 64;
  float acc = b3[0];
#pragma unroll 8
  for (int c = 0; c < 64; ++c) acc += (float)t[c] * sw[c];
  out[e] = 1.0f / (1.0f + expf(-acc));
}

// ---------------------------------------------------------------------------
// Host orchestration
// ---------------------------------------------------------------------------
extern "C" void kernel_launch(void* const* d_in, const int* in_sizes, int n_in,
                              void* d_out, int out_size, void* d_ws, size_t ws_size,
                              hipStream_t stream) {
  const float* x          = (const float*)d_in[0];
  const int*   edge_index = (const int*)d_in[1];
  const float* edge_attr  = (const float*)d_in[2];
  const int*   batch      = (const int*)d_in[3];
  const float* ne_w1 = (const float*)d_in[4];  const float* ne_b1 = (const float*)d_in[5];
  const float* ne_w2 = (const float*)d_in[6];  const float* ne_b2 = (const float*)d_in[7];
  const float* ne_g  = (const float*)d_in[8];  const float* ne_be = (const float*)d_in[9];
  const float* ee_w1 = (const float*)d_in[10]; const float* ee_b1 = (const float*)d_in[11];
  const float* ee_w2 = (const float*)d_in[12]; const float* ee_b2 = (const float*)d_in[13];
  const float* ee_g  = (const float*)d_in[14]; const float* ee_be = (const float*)d_in[15];
  const float* gin0_eps = (const float*)d_in[16];
  const float* gin0_w1 = (const float*)d_in[17]; const float* gin0_b1 = (const float*)d_in[18];
  const float* gin0_w2 = (const float*)d_in[19]; const float* gin0_b2 = (const float*)d_in[20];
  const float* gin0_g  = (const float*)d_in[21]; const float* gin0_be = (const float*)d_in[22];
  const float* gin1_eps = (const float*)d_in[23];
  const float* gin1_w1 = (const float*)d_in[24]; const float* gin1_b1 = (const float*)d_in[25];
  const float* gin1_w2 = (const float*)d_in[26]; const float* gin1_b2 = (const float*)d_in[27];
  const float* gin1_g  = (const float*)d_in[28]; const float* gin1_be = (const float*)d_in[29];
  const float* gp_w = (const float*)d_in[30]; const float* gp_b  = (const float*)d_in[31];
  const float* gp_g = (const float*)d_in[32]; const float* gp_be = (const float*)d_in[33];
  const float* ep_w1 = (const float*)d_in[34]; const float* ep_b1 = (const float*)d_in[35];
  const float* ep_w2 = (const float*)d_in[36]; const float* ep_b2 = (const float*)d_in[37];
  const float* ep_w3 = (const float*)d_in[38]; const float* ep_b3 = (const float*)d_in[39];

  const int N = in_sizes[0] / 3;
  const int E = in_sizes[1] / 2;
  const int G = 64;
  const int* src = edge_index;
  const int* dst = edge_index + E;

  // Workspace carve-out
  char* wsp = (char*)d_ws;
  auto alloc = [&](size_t bytes) -> void* {
    void* p = (void*)wsp;
    wsp += (bytes + 255) & ~(size_t)255;
    return p;
  };
  float*  h    = (float*)alloc((size_t)N * 64 * sizeof(float));
  float*  tmpN = (float*)alloc((size_t)N * 64 * sizeof(float));
  float*  aggr = (float*)alloc((size_t)N * 64 * sizeof(float));
  float*  ef   = (float*)alloc((size_t)E * 64 * sizeof(float));
  float*  tmpE = (float*)alloc((size_t)E * 64 * sizeof(float));   // reused for t2 (bf16)
  __bf16* Cmat = (__bf16*)alloc((size_t)E * 256 * sizeof(__bf16));
  __bf16* t1   = (__bf16*)alloc((size_t)E * 128 * sizeof(__bf16));
  __bf16* t2   = (__bf16*)tmpE;
  float*  seg  = (float*)alloc((size_t)(G * 64 + G) * sizeof(float));
  float*  cnt  = seg + G * 64;
  float*  gp   = (float*)alloc((size_t)G * 64 * sizeof(float));
  float*  gf   = (float*)alloc((size_t)G * 64 * sizeof(float));

  auto cdiv = [](long long a, long long b) -> unsigned { return (unsigned)((a + b - 1) / b); };

  // --- node encoder ---
  encode3_kernel<<<cdiv((long long)N * 64, 256), 256, 0, stream>>>(x, ne_w1, ne_b1, tmpN, N);
  gemm64_kernel<2><<<cdiv(N, 64), 128, 0, stream>>>(tmpN, nullptr, nullptr, ne_w2, ne_b2, ne_g, ne_be, h, N);
  // --- edge encoder ---
  encode3_kernel<<<cdiv((long long)E * 64, 256), 256, 0, stream>>>(edge_attr, ee_w1, ee_b1, tmpE, E);
  gemm64_kernel<2><<<cdiv(E, 64), 128, 0, stream>>>(tmpE, nullptr, nullptr, ee_w2, ee_b2, ee_g, ee_be, ef, E);

  // --- GIN layer 0 ---
  zero_kernel<<<cdiv((long long)N * 64, 256), 256, 0, stream>>>(aggr, (long long)N * 64);
  scatter_msg_kernel<<<cdiv((long long)E * 64, 256), 256, 0, stream>>>(h, ef, src, dst, aggr, E);
  gemm64_kernel<9><<<cdiv(N, 64), 128, 0, stream>>>(h, aggr, gin0_eps, gin0_w1, gin0_b1, nullptr, nullptr, tmpN, N);
  gemm64_kernel<6><<<cdiv(N, 64), 128, 0, stream>>>(tmpN, nullptr, nullptr, gin0_w2, gin0_b2, gin0_g, gin0_be, h, N);

  // --- GIN layer 1 ---
  zero_kernel<<<cdiv((long long)N * 64, 256), 256, 0, stream>>>(aggr, (long long)N * 64);
  scatter_msg_kernel<<<cdiv((long long)E * 64, 256), 256, 0, stream>>>(h, ef, src, dst, aggr, E);
  gemm64_kernel<9><<<cdiv(N, 64), 128, 0, stream>>>(h, aggr, gin1_eps, gin1_w1, gin1_b1, nullptr, nullptr, tmpN, N);
  gemm64_kernel<2><<<cdiv(N, 64), 128, 0, stream>>>(tmpN, nullptr, nullptr, gin1_w2, gin1_b2, gin1_g, gin1_be, h, N);

  // --- global mean pool + graph projection ---
  zero_kernel<<<cdiv(G * 64 + G, 256), 256, 0, stream>>>(seg, G * 64 + G);
  pool_scatter_kernel<<<cdiv((long long)N * 64, 256), 256, 0, stream>>>(h, batch, seg, cnt, N);
  pool_finalize_kernel<<<16, 256, 0, stream>>>(seg, cnt, gp);
  gemm64_kernel<3><<<1, 128, 0, stream>>>(gp, nullptr, nullptr, gp_w, gp_b, gp_g, gp_be, gf, G);

  // --- edge predictor ---
  concat_kernel<<<cdiv((long long)E * 4, 256), 256, 0, stream>>>(h, ef, gf, src, dst, batch, Cmat, E);
  gemm_big_kernel<256, 128><<<cdiv(E, 64), 128, 0, stream>>>(Cmat, ep_w1, ep_b1, t1, E);
  gemm_big_kernel<128, 64><<<cdiv(E, 64), 128, 0, stream>>>(t1, ep_w2, ep_b2, t2, E);
  final_kernel<<<cdiv(E, 256), 256, 0, stream>>>(t2, ep_w3, ep_b3, (float*)d_out, E);
}
